// CNNLSTMHybrid_19576460935473
// MI455X (gfx1250) — matrix operand order repaired and checked by reading, not compiled, not run on hardware
//
#include <hip/hip_runtime.h>
#include <hip/hip_bf16.h>

typedef __attribute__((ext_vector_type(16))) __bf16 v16bf;
typedef __attribute__((ext_vector_type(8)))  float  v8f;

#define B_     8
#define S_     1024
#define T_     256
#define E_     256
#define H_     512
#define SENC_  256
#define VOCAB_ 10

// ---------------------------------------------------------------------------
// Generic batched WMMA GEMM:  C[z] = A[z] (MxK) * Bt[z]^T (N x K) + bias, with
// optional "shifts" (conv taps: shift s advances A by s rows and Bt by s*N*K).
// A, Bt are bf16 row-major.  One wave per 16x16 C tile.
// Fragment layouts follow CDNA5 ISA 7.12.2 (wave32):
//   A 16x32 bf16 : lane(m=lane&15, half=lane>>4), elem j<8 -> K=half*8+j,
//                  elem j>=8 -> K=16+half*8+(j-8)    (two contiguous 8-runs)
//   B 32x16 bf16 : lane(n=lane&15, half), elem j -> same K as A elem j (two 8-runs
//                  of the N x K row-major weight row)
//   C/D f32 16x16: lane(n=lane&15), vgpr r -> M=r+half*8
// ---------------------------------------------------------------------------
__global__ __launch_bounds__(32)
void gemm_wmma_bf16(const __bf16* __restrict__ A, const __bf16* __restrict__ Bt,
                    const float* __restrict__ bias,
                    float* __restrict__ Cf, __bf16* __restrict__ Cb,
                    int M, int N, int K,
                    int lda, long long strideAB, long long strideBB,
                    int ldc, long long strideCB,
                    int shifts, int relu)
{
    const int lane  = threadIdx.x;
    const int col16 = lane & 15;
    const int half  = lane >> 4;
    const int m0 = blockIdx.y * 16;
    const int n0 = blockIdx.x * 16;

    const __bf16* Ab = A  + (long long)blockIdx.z * strideAB + (long long)m0 * lda;
    const __bf16* Bb = Bt + (long long)blockIdx.z * strideBB;

    v8f acc;
    const float bv = bias ? bias[n0 + col16] : 0.0f;
#pragma unroll
    for (int r = 0; r < 8; ++r) acc[r] = bv;

    for (int s = 0; s < shifts; ++s) {
        const __bf16* As = Ab + (long long)s * lda;            // conv tap shift
        const __bf16* Bs = Bb + (long long)s * (long long)N * K;
        for (int k0 = 0; k0 < K; k0 += 32) {
            v16bf a, b;
            const __bf16* ap = As + (long long)col16 * lda + k0 + half * 8;
#pragma unroll
            for (int j = 0; j < 8; ++j) a[j]     = ap[j];
#pragma unroll
            for (int j = 0; j < 8; ++j) a[8 + j] = ap[16 + j];
            const __bf16* bp = Bs + (long long)(n0 + col16) * K + k0 + half * 8;
#pragma unroll
            for (int j = 0; j < 16; ++j) b[j] = bp[j + (j >> 3) * 8];
            acc = __builtin_amdgcn_wmma_f32_16x16x32_bf16(
                false, a, false, b, (short)0, acc, false, false);
        }
    }

    const int col = n0 + col16;
    const int rb  = half * 8;
#pragma unroll
    for (int r = 0; r < 8; ++r) {
        float v = acc[r];
        if (relu) v = fmaxf(v, 0.0f);
        const long long off =
            (long long)blockIdx.z * strideCB + (long long)(m0 + rb + r) * ldc + col;
        if (Cf) Cf[off] = v;
        if (Cb) Cb[off] = (__bf16)v;
    }
}

// ---------------------------------------------------------------------------
// Persistent LSTM recurrence: one 1024-thread workgroup (32 waves).
// Wave w owns hidden cols [16w, 16w+16) and computes its 4 gate tiles
// (i,f,g,o) with WMMA each step; cell state lives in the accumulator layout
// in registers; h round-trips through LDS (rows 8..15 zero-padded, M=8 batch).
// xg = x @ wih^T + (bih + bhh) precomputed by the big GEMM, layout (B*T, 4H).
// whh is bf16 (4H x H) row-major = N x K, gate order i,f,g,o.
// ---------------------------------------------------------------------------
__global__ __launch_bounds__(1024, 1)
void lstm_recurrence(const float* __restrict__ xg, const __bf16* __restrict__ whh,
                     float* __restrict__ hf, __bf16* __restrict__ hb, int T)
{
    __shared__ __bf16 h_lds[16 * H_];

    const int tid   = threadIdx.x;
    const int wave  = tid >> 5;          // 0..31
    const int lane  = tid & 31;
    const int col16 = lane & 15;
    const int half  = lane >> 4;
    const int hcol  = wave * 16 + col16; // hidden unit 0..511

    for (int i = tid; i < 16 * H_; i += 1024) h_lds[i] = (__bf16)0.0f;
    v8f c = {};
    __syncthreads();

    for (int t = 0; t < T; ++t) {
        v8f a0 = {}, a1 = {}, a2 = {}, a3 = {};
        if (half == 0) {                 // rows 0..7 == batch 0..7
#pragma unroll
            for (int r = 0; r < 8; ++r) {
                const long long base = ((long long)r * T + t) * (4 * H_);
                a0[r] = xg[base + 0 * H_ + hcol];
                a1[r] = xg[base + 1 * H_ + hcol];
                a2[r] = xg[base + 2 * H_ + hcol];
                a3[r] = xg[base + 3 * H_ + hcol];
            }
        }
        for (int k0 = 0; k0 < H_; k0 += 32) {
            v16bf a;
            {
                const int kb = k0 + half * 8;
                const __bf16* hp = &h_lds[col16 * H_ + kb];
#pragma unroll
                for (int j = 0; j < 8; ++j) a[j]     = hp[j];
#pragma unroll
                for (int j = 0; j < 8; ++j) a[8 + j] = hp[16 + j];
            }
            const int kb = k0 + half * 8;
            const __bf16* w0 = whh + (long long)(0 * H_ + hcol) * H_ + kb;
            const __bf16* w1 = whh + (long long)(1 * H_ + hcol) * H_ + kb;
            const __bf16* w2 = whh + (long long)(2 * H_ + hcol) * H_ + kb;
            const __bf16* w3 = whh + (long long)(3 * H_ + hcol) * H_ + kb;
            v16bf b0, b1, b2, b3;
#pragma unroll
            for (int j = 0; j < 16; ++j) { const int jj = j + (j >> 3) * 8;
                                           b0[j] = w0[jj]; b1[j] = w1[jj];
                                           b2[j] = w2[jj]; b3[j] = w3[jj]; }
            a0 = __builtin_amdgcn_wmma_f32_16x16x32_bf16(false, a, false, b0, (short)0, a0, false, false);
            a1 = __builtin_amdgcn_wmma_f32_16x16x32_bf16(false, a, false, b1, (short)0, a1, false, false);
            a2 = __builtin_amdgcn_wmma_f32_16x16x32_bf16(false, a, false, b2, (short)0, a2, false, false);
            a3 = __builtin_amdgcn_wmma_f32_16x16x32_bf16(false, a, false, b3, (short)0, a3, false, false);
        }
        __syncthreads();                 // all h reads for step t done
        if (half == 0) {
#pragma unroll
            for (int r = 0; r < 8; ++r) {
                const float ig = 1.0f / (1.0f + __expf(-a0[r]));
                const float fg = 1.0f / (1.0f + __expf(-a1[r]));
                const float gg = tanhf(a2[r]);
                const float og = 1.0f / (1.0f + __expf(-a3[r]));
                const float cn = fg * c[r] + ig * gg;
                c[r] = cn;
                const float hv = og * tanhf(cn);
                h_lds[r * H_ + hcol] = (__bf16)hv;
                const long long o = ((long long)r * T + t) * H_ + hcol;
                hf[o] = hv;
                if (hb) hb[o] = (__bf16)hv;
            }
        }
        __syncthreads();                 // new h visible for step t+1
    }
}

// ------------------------- small support kernels ---------------------------
__global__ void embed_src_pad(const int* __restrict__ src, const float* __restrict__ emb,
                              __bf16* __restrict__ out)
{   // out: (B, S+2, E), zero halo rows 0 and S+1
    long long idx = (long long)blockIdx.x * blockDim.x + threadIdx.x;
    const long long total = (long long)B_ * (S_ + 2) * E_;
    if (idx >= total) return;
    const int e   = (int)(idx % E_);
    long long rst = idx / E_;
    const int row = (int)(rst % (S_ + 2));
    const int b   = (int)(rst / (S_ + 2));
    float v = 0.0f;
    if (row >= 1 && row <= S_) v = emb[(long long)src[b * S_ + row - 1] * E_ + e];
    out[idx] = (__bf16)v;
}

__global__ void embed_tgt(const int* __restrict__ tgt, const float* __restrict__ emb,
                          __bf16* __restrict__ out)
{   // out: (B*T, E)
    long long idx = (long long)blockIdx.x * blockDim.x + threadIdx.x;
    const long long total = (long long)B_ * T_ * E_;
    if (idx >= total) return;
    const int e = (int)(idx % E_);
    const int r = (int)(idx / E_);
    out[idx] = (__bf16)emb[(long long)tgt[r] * E_ + e];
}

__global__ void pool_pad(const float* __restrict__ in, __bf16* __restrict__ out,
                         int L, int C)
{   // in (B, L, C) relu'd -> out (B, L/2+2, C) bf16, zero halo rows
    const int Lo = L / 2 + 2;
    long long idx = (long long)blockIdx.x * blockDim.x + threadIdx.x;
    const long long total = (long long)B_ * Lo * C;
    if (idx >= total) return;
    const int ch  = (int)(idx % C);
    long long rst = idx / C;
    const int row = (int)(rst % Lo);
    const int b   = (int)(rst / Lo);
    float v = 0.0f;
    if (row >= 1 && row <= L / 2) {
        const int p = row - 1;
        const float x0 = in[((long long)b * L + 2 * p) * C + ch];
        const float x1 = in[((long long)b * L + 2 * p + 1) * C + ch];
        v = fmaxf(x0, x1);
    }
    out[idx] = (__bf16)v;
}

__global__ void conv_w_reorg(const float* __restrict__ w, __bf16* __restrict__ out,
                             int Cout, int Cin)
{   // (Cout, Cin, 3) f32 -> (3, Cout, Cin) bf16
    long long idx = (long long)blockIdx.x * blockDim.x + threadIdx.x;
    const long long total = 3LL * Cout * Cin;
    if (idx >= total) return;
    const int ci  = (int)(idx % Cin);
    long long rst = idx / Cin;
    const int co  = (int)(rst % Cout);
    const int k   = (int)(rst / Cout);
    out[idx] = (__bf16)w[((long long)co * Cin + ci) * 3 + k];
}

__global__ void f32_to_bf16(const float* __restrict__ a, __bf16* __restrict__ o, long long n)
{
    long long i = (long long)blockIdx.x * blockDim.x + threadIdx.x;
    if (i < n) o[i] = (__bf16)a[i];
}

__global__ void vec_add(const float* __restrict__ a, const float* __restrict__ b,
                        float* __restrict__ o, int n)
{
    int i = blockIdx.x * blockDim.x + threadIdx.x;
    if (i < n) o[i] = a[i] + b[i];
}

__global__ void transpose_bf16(const float* __restrict__ in, __bf16* __restrict__ out,
                               int R, int C)
{   // (B, R, C) f32 -> (B, C, R) bf16
    long long idx = (long long)blockIdx.x * blockDim.x + threadIdx.x;
    const long long total = (long long)B_ * R * C;
    if (idx >= total) return;
    const int r   = (int)(idx % R);
    long long rst = idx / R;
    const int c   = (int)(rst % C);
    const int b   = (int)(rst / C);
    out[idx] = (__bf16)in[((long long)b * R + r) * C + c];
}

__global__ void rowdot(const float* __restrict__ X, const float* __restrict__ w,
                       float* __restrict__ out, int rows, int K)
{
    int r = blockIdx.x * blockDim.x + threadIdx.x;
    if (r >= rows) return;
    float s = 0.0f;
    for (int k = 0; k < K; ++k) s += X[(long long)r * K + k] * w[k];
    out[r] = s;
}

__global__ __launch_bounds__(32)
void attn_softmax(const float* __restrict__ dec_p, const float* __restrict__ enc_p,
                  const float* __restrict__ attn_b, __bf16* __restrict__ attn)
{   // one wave per (b,t) row; Senc = 256 -> 8 entries per lane
    const int row  = blockIdx.x;          // b*T + t
    const int b    = row >> 8;            // T == 256
    const int lane = threadIdx.x;
    const float dp = dec_p[row] + attn_b[0];
    float sc[8];
    float mx = -3.0e38f;
#pragma unroll
    for (int j = 0; j < 8; ++j) {
        const float v = dp + enc_p[b * SENC_ + lane + 32 * j];
        sc[j] = v;
        mx = fmaxf(mx, v);
    }
#pragma unroll
    for (int o = 16; o > 0; o >>= 1) mx = fmaxf(mx, __shfl_xor(mx, o, 32));
    float sum = 0.0f;
#pragma unroll
    for (int j = 0; j < 8; ++j) { sc[j] = __expf(sc[j] - mx); sum += sc[j]; }
#pragma unroll
    for (int o = 16; o > 0; o >>= 1) sum += __shfl_xor(sum, o, 32);
    const float inv = 1.0f / sum;
#pragma unroll
    for (int j = 0; j < 8; ++j)
        attn[(long long)row * SENC_ + lane + 32 * j] = (__bf16)(sc[j] * inv);
}

__global__ void out_proj(const float* __restrict__ dec, const float* __restrict__ ow,
                         const float* __restrict__ ob, float* __restrict__ out, int rows)
{   // (rows, H) @ (VOCAB, H)^T + b
    int idx = blockIdx.x * blockDim.x + threadIdx.x;
    if (idx >= rows * VOCAB_) return;
    const int v = idx % VOCAB_;
    const int r = idx / VOCAB_;
    float s = ob[v];
    for (int k = 0; k < H_; ++k) s += dec[(long long)r * H_ + k] * ow[v * H_ + k];
    out[idx] = s;
}

// ---------------------------------------------------------------------------
extern "C" void kernel_launch(void* const* d_in, const int* in_sizes, int n_in,
                              void* d_out, int out_size, void* d_ws, size_t ws_size,
                              hipStream_t stream)
{
    (void)in_sizes; (void)n_in; (void)out_size; (void)ws_size;

    const int*   src   = (const int*)  d_in[0];
    const int*   tgt   = (const int*)  d_in[1];
    const float* emb   = (const float*)d_in[2];
    const float* c1w   = (const float*)d_in[3];
    const float* c1b   = (const float*)d_in[4];
    const float* c2w   = (const float*)d_in[5];
    const float* c2b   = (const float*)d_in[6];
    const float* c3w   = (const float*)d_in[7];
    const float* c3b   = (const float*)d_in[8];
    const float* wih0  = (const float*)d_in[9];
    const float* whh0  = (const float*)d_in[10];
    const float* bih0  = (const float*)d_in[11];
    const float* bhh0  = (const float*)d_in[12];
    const float* wih1  = (const float*)d_in[13];
    const float* whh1  = (const float*)d_in[14];
    const float* bih1  = (const float*)d_in[15];
    const float* bhh1  = (const float*)d_in[16];
    const float* attnw = (const float*)d_in[17];
    const float* attnb = (const float*)d_in[18];
    const float* outw  = (const float*)d_in[19];
    const float* outb  = (const float*)d_in[20];
    float* out = (float*)d_out;                    // [B*T*VOCAB | B*T*H]
    float* ctx = out + (long long)B_ * T_ * VOCAB_;

    size_t off = 0;
    auto alloc = [&](size_t bytes) -> void* {
        off = (off + 255) & ~(size_t)255;
        void* p = (char*)d_ws + off;
        off += bytes;
        return p;
    };
    __bf16* xsrc   = (__bf16*)alloc((size_t)B_ * (S_ + 2) * E_ * 2);   // padded src embed
    __bf16* xtgt   = (__bf16*)alloc((size_t)B_ * T_ * E_ * 2);
    float*  conv1o = (float*) alloc((size_t)B_ * S_ * 256 * 4);
    __bf16* pool1  = (__bf16*)alloc((size_t)B_ * (S_/2 + 2) * 256 * 2);
    float*  conv2o = (float*) alloc((size_t)B_ * (S_/2) * 512 * 4);
    __bf16* pool2  = (__bf16*)alloc((size_t)B_ * (S_/4 + 2) * 512 * 2);
    float*  enc    = (float*) alloc((size_t)B_ * SENC_ * H_ * 4);
    __bf16* encT   = (__bf16*)alloc((size_t)B_ * H_ * SENC_ * 2);
    __bf16* w1b    = (__bf16*)alloc((size_t)3 * 256 * E_ * 2);
    __bf16* w2b    = (__bf16*)alloc((size_t)3 * 512 * 256 * 2);
    __bf16* w3b    = (__bf16*)alloc((size_t)3 * 512 * 512 * 2);
    __bf16* wih0b  = (__bf16*)alloc((size_t)4 * H_ * E_ * 2);
    __bf16* whh0b  = (__bf16*)alloc((size_t)4 * H_ * H_ * 2);
    __bf16* wih1b  = (__bf16*)alloc((size_t)4 * H_ * H_ * 2);
    __bf16* whh1b  = (__bf16*)alloc((size_t)4 * H_ * H_ * 2);
    float*  bias0  = (float*) alloc((size_t)4 * H_ * 4);
    float*  bias1  = (float*) alloc((size_t)4 * H_ * 4);
    float*  xg0    = (float*) alloc((size_t)B_ * T_ * 4 * H_ * 4);
    float*  xg1    = (float*) alloc((size_t)B_ * T_ * 4 * H_ * 4);
    float*  dec0f  = (float*) alloc((size_t)B_ * T_ * H_ * 4);
    __bf16* dec0b  = (__bf16*)alloc((size_t)B_ * T_ * H_ * 2);
    float*  dec1f  = (float*) alloc((size_t)B_ * T_ * H_ * 4);
    float*  dec_p  = (float*) alloc((size_t)B_ * T_ * 4);
    float*  enc_p  = (float*) alloc((size_t)B_ * SENC_ * 4);
    __bf16* attn   = (__bf16*)alloc((size_t)B_ * T_ * SENC_ * 2);

    const dim3 blk(256);
    auto gs = [](long long n) { return dim3((unsigned)((n + 255) / 256)); };

    // --- weight prep ---
    hipLaunchKernelGGL(conv_w_reorg, gs(3LL*256*E_),  blk, 0, stream, c1w, w1b, 256, E_);
    hipLaunchKernelGGL(conv_w_reorg, gs(3LL*512*256), blk, 0, stream, c2w, w2b, 512, 256);
    hipLaunchKernelGGL(conv_w_reorg, gs(3LL*512*512), blk, 0, stream, c3w, w3b, 512, 512);
    hipLaunchKernelGGL(f32_to_bf16, gs(4LL*H_*E_), blk, 0, stream, wih0, wih0b, 4LL*H_*E_);
    hipLaunchKernelGGL(f32_to_bf16, gs(4LL*H_*H_), blk, 0, stream, whh0, whh0b, 4LL*H_*H_);
    hipLaunchKernelGGL(f32_to_bf16, gs(4LL*H_*H_), blk, 0, stream, wih1, wih1b, 4LL*H_*H_);
    hipLaunchKernelGGL(f32_to_bf16, gs(4LL*H_*H_), blk, 0, stream, whh1, whh1b, 4LL*H_*H_);
    hipLaunchKernelGGL(vec_add, gs(4*H_), blk, 0, stream, bih0, bhh0, bias0, 4*H_);
    hipLaunchKernelGGL(vec_add, gs(4*H_), blk, 0, stream, bih1, bhh1, bias1, 4*H_);

    // --- embeddings ---
    hipLaunchKernelGGL(embed_src_pad, gs((long long)B_*(S_+2)*E_), blk, 0, stream, src, emb, xsrc);
    hipLaunchKernelGGL(embed_tgt,     gs((long long)B_*T_*E_),     blk, 0, stream, tgt, emb, xtgt);

    // --- CNN encoder: conv as 3 shifted WMMA GEMMs, fused bias+ReLU ---
    hipLaunchKernelGGL(gemm_wmma_bf16, dim3(256/16, S_/16, B_), dim3(32), 0, stream,
        xsrc, w1b, c1b, conv1o, (__bf16*)nullptr,
        S_, 256, E_, E_, (long long)(S_+2)*E_, 0LL, 256, (long long)S_*256, 3, 1);
    hipLaunchKernelGGL(pool_pad, gs((long long)B_*(S_/2+2)*256), blk, 0, stream, conv1o, pool1, S_, 256);

    hipLaunchKernelGGL(gemm_wmma_bf16, dim3(512/16, (S_/2)/16, B_), dim3(32), 0, stream,
        pool1, w2b, c2b, conv2o, (__bf16*)nullptr,
        S_/2, 512, 256, 256, (long long)(S_/2+2)*256, 0LL, 512, (long long)(S_/2)*512, 3, 1);
    hipLaunchKernelGGL(pool_pad, gs((long long)B_*(S_/4+2)*512), blk, 0, stream, conv2o, pool2, S_/2, 512);

    hipLaunchKernelGGL(gemm_wmma_bf16, dim3(512/16, SENC_/16, B_), dim3(32), 0, stream,
        pool2, w3b, c3b, enc, (__bf16*)nullptr,
        SENC_, 512, 512, 512, (long long)(SENC_+2)*512, 0LL, 512, (long long)SENC_*512, 3, 1);
    hipLaunchKernelGGL(transpose_bf16, gs((long long)B_*H_*SENC_), blk, 0, stream, enc, encT, SENC_, H_);

    // --- LSTM decoder ---
    // layer 0 input projection: (B*T, E) @ (4H, E)^T + bias
    hipLaunchKernelGGL(gemm_wmma_bf16, dim3(4*H_/16, (B_*T_)/16, 1), dim3(32), 0, stream,
        xtgt, wih0b, bias0, xg0, (__bf16*)nullptr,
        B_*T_, 4*H_, E_, E_, 0LL, 0LL, 4*H_, 0LL, 1, 0);
    hipLaunchKernelGGL(lstm_recurrence, dim3(1), dim3(1024), 0, stream, xg0, whh0b, dec0f, dec0b, T_);

    // layer 1 input projection: (B*T, H) @ (4H, H)^T + bias
    hipLaunchKernelGGL(gemm_wmma_bf16, dim3(4*H_/16, (B_*T_)/16, 1), dim3(32), 0, stream,
        dec0b, wih1b, bias1, xg1, (__bf16*)nullptr,
        B_*T_, 4*H_, H_, H_, 0LL, 0LL, 4*H_, 0LL, 1, 0);
    hipLaunchKernelGGL(lstm_recurrence, dim3(1), dim3(1024), 0, stream, xg1, whh1b, dec1f, (__bf16*)nullptr, T_);

    // --- attention ---
    hipLaunchKernelGGL(rowdot, gs(B_*T_),    blk, 0, stream, dec1f, attnw,      dec_p, B_*T_,    H_);
    hipLaunchKernelGGL(rowdot, gs(B_*SENC_), blk, 0, stream, enc,   attnw + H_, enc_p, B_*SENC_, H_);
    hipLaunchKernelGGL(attn_softmax, dim3(B_*T_), dim3(32), 0, stream, dec_p, enc_p, attnb, attn);

    // context = attn (T x Senc) @ encT (H x Senc)^T, batched over B -> d_out ctx
    hipLaunchKernelGGL(gemm_wmma_bf16, dim3(H_/16, T_/16, B_), dim3(32), 0, stream,
        attn, encT, (const float*)nullptr, ctx, (__bf16*)nullptr,
        T_, H_, SENC_, SENC_, (long long)T_*SENC_, (long long)H_*SENC_,
        H_, (long long)T_*H_, 1, 0);

    // --- vocab projection (N=10, scalar) ---
    hipLaunchKernelGGL(out_proj, gs((long long)B_*T_*VOCAB_), blk, 0, stream, dec1f, outw, outb, out, B_*T_);
}
